// TimeFilterBackbone_65309272703219
// MI455X (gfx1250) — matrix operand order repaired and checked by reading, not compiled, not run on hardware
//
#include <hip/hip_runtime.h>
#include <hip/hip_fp16.h>

#define Bv   2
#define Lv   1408
#define Dv   512
#define Hv   8
#define DHv  64
#define DFFv 2048
#define BHv  16      // B*H
#define LTv  88      // L/16
#define KSEL 704     // int(0.5*L)
#define KCH  512     // TDM A-panel chunk (halves)

typedef __attribute__((ext_vector_type(16))) _Float16 v16h;
typedef __attribute__((ext_vector_type(8)))  float    v8f;
typedef __attribute__((ext_vector_type(8)))  unsigned v8u;
typedef unsigned u32x4 __attribute__((ext_vector_type(4)));
typedef int      i32x8 __attribute__((ext_vector_type(8)));
typedef int      i32x4 __attribute__((ext_vector_type(4)));

// ---------------- WMMA fragment loaders (CDNA5 wave32 layouts, dword loads) ----------------
// A 16x32 f16: lane m=lane&15, half=lane>>4; VGPR v: K = (v/4)*16 + half*8 + (v%4)*2
__device__ __forceinline__ v16h load_a_frag(const _Float16* __restrict__ A, int lda) {
  int lane = threadIdx.x & 31;
  int m = lane & 15, hl = lane >> 4;
  union { v8u u; v16h h; } r;
#pragma unroll
  for (int v = 0; v < 8; ++v) {
    int k = ((v >> 2) << 4) + (hl << 3) + ((v & 3) << 1);
    r.u[v] = *(const unsigned*)(A + (size_t)m * lda + k);   // (k,k+1) pair
  }
  return r.h;
}
// B 32x16 from TRANSPOSED storage Bt[N,K]: B(k,n) = Bt[n*ldt + k]; lane n, K = (v/4)*16 + half*8 + (v%4)*2
__device__ __forceinline__ v16h load_bT_frag(const _Float16* __restrict__ Bt, size_t ldt) {
  int lane = threadIdx.x & 31;
  int n = lane & 15, hl = lane >> 4;
  union { v8u u; v16h h; } r;
#pragma unroll
  for (int v = 0; v < 8; ++v) {
    int k = ((v >> 2) << 4) + (hl << 3) + ((v & 3) << 1);
    r.u[v] = *(const unsigned*)(Bt + (size_t)n * ldt + k);  // (k,k+1) pair
  }
  return r.h;
}

__device__ __forceinline__ float gelu_exact(float v) {
  return 0.5f * v * (1.0f + erff(v * 0.70710678118654752f));
}

// ---------------- small helpers ----------------
__global__ void cvt_f16(const float* __restrict__ s, _Float16* __restrict__ d, int n) {
  int i = blockIdx.x * 256 + threadIdx.x;
  if (i < n) d[i] = (_Float16)s[i];
}
// f32 [K,N] -> f16 transposed [N,K]
__global__ void cvt_f16_t(const float* __restrict__ s, _Float16* __restrict__ d, int K, int N) {
  int i = blockIdx.x * 256 + threadIdx.x;
  if (i < K * N) {
    int k = i / N, n = i % N;
    d[(size_t)n * K + k] = (_Float16)s[i];
  }
}

// LayerNorm over D=512 per row, write f16
__global__ void ln_kernel(const float* __restrict__ x, const float* __restrict__ g,
                          const float* __restrict__ b, _Float16* __restrict__ o16) {
  __shared__ float r1[256], r2[256];
  int row = blockIdx.x, tid = threadIdx.x;
  const float* xr = x + (size_t)row * Dv;
  float a0 = xr[tid], a1 = xr[tid + 256];
  r1[tid] = a0 + a1; r2[tid] = a0 * a0 + a1 * a1;
  __syncthreads();
  for (int s = 128; s > 0; s >>= 1) {
    if (tid < s) { r1[tid] += r1[tid + s]; r2[tid] += r2[tid + s]; }
    __syncthreads();
  }
  float m = r1[0] * (1.0f / Dv);
  float var = r2[0] * (1.0f / Dv) - m * m;
  float rstd = rsqrtf(var + 1e-5f);
  _Float16* orow = o16 + (size_t)row * Dv;
  orow[tid]       = (_Float16)((a0 - m) * rstd * g[tid]       + b[tid]);
  orow[tid + 256] = (_Float16)((a1 - m) * rstd * g[tid + 256] + b[tid + 256]);
}

// ---------------- per-head Q/K projections: [L,64] @ [64,64] + bias ----------------
__global__ void qk_proj_kernel(const _Float16* __restrict__ xn16,
                               const _Float16* __restrict__ wp1t, const float* __restrict__ bp1,
                               const _Float16* __restrict__ wp2t, const float* __restrict__ bp2,
                               _Float16* __restrict__ q16, _Float16* __restrict__ k16) {
  int wave = threadIdx.x >> 5;
  int which = wave >> 2, nt = wave & 3;
  int lt = blockIdx.x % LTv; int bh = blockIdx.x / LTv;
  int b = bh / Hv, h = bh % Hv;
  const _Float16* Ab = xn16 + ((size_t)b * Lv + lt * 16) * Dv + h * DHv;
  const _Float16* Wt = (which ? wp2t : wp1t) + (size_t)(nt * 16) * DHv;  // [N,K]
  const float* bias  = which ? bp2 : bp1;
  v8f acc = {};
#pragma unroll
  for (int k0 = 0; k0 < DHv; k0 += 32) {
    v16h a  = load_a_frag(Ab + k0, Dv);
    v16h bf = load_bT_frag(Wt + k0, DHv);
    acc = __builtin_amdgcn_wmma_f32_16x16x32_f16(false, a, false, bf, (short)0, acc, false, false);
  }
  _Float16* O = (which ? k16 : q16) + ((size_t)bh * Lv + lt * 16) * DHv + nt * 16;
  int lane = threadIdx.x & 31; int n = lane & 15, hl = lane >> 4;
#pragma unroll
  for (int r = 0; r < 8; ++r)
    O[(size_t)(hl * 8 + r) * DHv + n] = (_Float16)(acc[r] + bias[nt * 16 + n]);
}

// ---------------- adj = gelu(Q @ K^T), per (b,h): [L,64]x[64,L] -> f32 ----------------
__global__ void adj_gemm_kernel(const _Float16* __restrict__ q16, const _Float16* __restrict__ k16,
                                float* __restrict__ adj) {
  int wave = threadIdx.x >> 5;
  int cb = blockIdx.x % 11; int t = blockIdx.x / 11;
  int rt = t % LTv; int bh = t / LTv;
  int ct = cb * 8 + wave;
  const _Float16* Qb = q16 + ((size_t)bh * Lv + rt * 16) * DHv;
  const _Float16* Kb = k16 + ((size_t)bh * Lv + ct * 16) * DHv;  // K rows = columns of B
  v8f acc = {};
#pragma unroll
  for (int k0 = 0; k0 < DHv; k0 += 32) {
    v16h a  = load_a_frag(Qb + k0, DHv);
    v16h bf = load_bT_frag(Kb + k0, DHv);
    acc = __builtin_amdgcn_wmma_f32_16x16x32_f16(false, a, false, bf, (short)0, acc, false, false);
  }
  float* O = adj + ((size_t)bh * Lv + rt * 16) * Lv + ct * 16;
  int lane = threadIdx.x & 31; int n = lane & 15, hl = lane >> 4;
#pragma unroll
  for (int r = 0; r < 8; ++r)
    O[(size_t)(hl * 8 + r) * Lv + n] = gelu_exact(acc[r]);
}

// ---------------- mask_topk: zero the 704 smallest per row (earliest-index ties) ----------------
__global__ void topk_kernel(float* __restrict__ adj) {
  __shared__ unsigned su[Lv];
  __shared__ int red[256];
  __shared__ unsigned s_pref;
  __shared__ int s_kk, s_less;
  int tid = threadIdx.x;
  float* ar = adj + (size_t)blockIdx.x * Lv;
  for (int j = tid; j < Lv; j += 256) {
    unsigned u = __float_as_uint(ar[j]);
    su[j] = (u & 0x80000000u) ? ~u : (u | 0x80000000u);   // order-preserving map
  }
  if (tid == 0) { s_pref = 0u; s_kk = KSEL; }
  __syncthreads();
  unsigned msk = 0u;
  for (int bit = 31; bit >= 0; --bit) {                    // radix select: k-th smallest
    unsigned bi = 1u << bit, nm = msk | bi, pref = s_pref;
    int c = 0;
    for (int j = tid; j < Lv; j += 256) c += ((su[j] & nm) == pref) ? 1 : 0;
    red[tid] = c; __syncthreads();
    for (int s = 128; s > 0; s >>= 1) { if (tid < s) red[tid] += red[tid + s]; __syncthreads(); }
    if (tid == 0) { if (red[0] < s_kk) { s_pref = pref | bi; s_kk -= red[0]; } }
    __syncthreads();
    msk = nm;
  }
  unsigned t = s_pref;
  int c = 0;
  for (int j = tid; j < Lv; j += 256) c += (su[j] < t) ? 1 : 0;
  red[tid] = c; __syncthreads();
  for (int s = 128; s > 0; s >>= 1) { if (tid < s) red[tid] += red[tid + s]; __syncthreads(); }
  if (tid == 0) s_less = red[0];
  __syncthreads();
  int rEq = KSEL - s_less;                                  // # of ==t to zero, earliest-index first
  const int CH = 6;
  int st = tid * CH; if (st > Lv) st = Lv;
  int en = st + CH;  if (en > Lv) en = Lv;
  int loc = 0;
  for (int j = st; j < en; ++j) loc += (su[j] == t) ? 1 : 0;
  red[tid] = loc; __syncthreads();
  for (int s = 1; s < 256; s <<= 1) {                       // inclusive scan
    int v = (tid >= s) ? red[tid - s] : 0;
    __syncthreads(); red[tid] += v; __syncthreads();
  }
  int rank = red[tid] - loc;
  for (int j = st; j < en; ++j) {
    unsigned u = su[j];
    if (u < t) ar[j] = 0.0f;
    else if (u == t) { if (rank < rEq) ar[j] = 0.0f; ++rank; }
  }
}

// ---------------- MoE gating + loss statistics (deterministic buffers) ----------------
__global__ void gate_kernel(const float* __restrict__ adj, const float* __restrict__ wg,
                            float* __restrict__ gates, float* __restrict__ sps,
                            float* __restrict__ ent) {
  __shared__ float r0[256], r1[256], r2[256];
  int row = blockIdx.x, tid = threadIdx.x;
  const float* ar = adj + (size_t)row * Lv;
  float a0 = 0, a1 = 0, a2 = 0;
  for (int j = tid; j < Lv; j += 256) {
    float a = ar[j];
    a0 += a * wg[j * 3 + 0]; a1 += a * wg[j * 3 + 1]; a2 += a * wg[j * 3 + 2];
  }
  r0[tid] = a0; r1[tid] = a1; r2[tid] = a2; __syncthreads();
  for (int s = 128; s > 0; s >>= 1) {
    if (tid < s) { r0[tid] += r0[tid + s]; r1[tid] += r1[tid + s]; r2[tid] += r2[tid + s]; }
    __syncthreads();
  }
  if (tid == 0) {
    float d0 = r0[0], d1 = r1[0], d2 = r2[0];
    float mx = fmaxf(d0, fmaxf(d1, d2));
    float e0 = expf(d0 - mx), e1 = expf(d1 - mx), e2 = expf(d2 - mx);
    float s = e0 + e1 + e2;
    float p[3] = { e0 / s, e1 / s, e2 / s };
    int ord[3] = { 0, 1, 2 };                              // stable descending sort
    for (int ps = 0; ps < 2; ++ps)
      for (int i = 0; i < 2 - ps; ++i)
        if (p[ord[i]] < p[ord[i + 1]]) { int tt = ord[i]; ord[i] = ord[i + 1]; ord[i + 1] = tt; }
    float sp0 = p[ord[0]], sp1 = p[ord[1]], sp2 = p[ord[2]];
    float c0 = sp0, c1 = sp0 + sp1, c2 = c1 + sp2;
    bool m0 = c0 > 0.5f, m1 = c1 > 0.5f, m2 = c2 > 0.5f;
    int thr = m0 ? 0 : (m1 ? 1 : 2);
    bool keep[3];
    keep[0] = !(m0 && thr != 0);
    keep[1] = !(m1 && thr != 1);
    keep[2] = !(m2 && thr != 2);
    float* g = gates + (size_t)row * 3;
    g[ord[0]] = keep[0] ? 1.f : 0.f;
    g[ord[1]] = keep[1] ? 1.f : 0.f;
    g[ord[2]] = keep[2] ? 1.f : 0.f;
    float* sp = sps + (size_t)row * 3;                     // kept sp in sorted space
    sp[0] = keep[0] ? sp0 : 0.f;
    sp[1] = keep[1] ? sp1 : 0.f;
    sp[2] = keep[2] ? sp2 : 0.f;
    ent[row] = -(p[0] * logf(p[0] + 1e-10f) + p[1] * logf(p[1] + 1e-10f) + p[2] * logf(p[2] + 1e-10f));
  }
}

// ---------------- final_mask mix + softmax; overlay f16 row in-place ----------------
__global__ void fsm_kernel(float* __restrict__ adj, const float* __restrict__ gates,
                           const float* __restrict__ masks) {
  __shared__ float srow[Lv];
  __shared__ float red[256];
  int row = blockIdx.x, tid = threadIdx.x;
  int l = row % Lv;
  const float* g = gates + (size_t)row * 3;
  float g0 = g[0], g1 = g[1], g2 = g[2];
  const float* m0 = masks + (size_t)l * 3 * Lv;
  const float* m1 = m0 + Lv;
  const float* m2 = m1 + Lv;
  float* ar = adj + (size_t)row * Lv;
  float mx = -3.4e38f;
  for (int j = tid; j < Lv; j += 256) {
    float fm = g0 * m0[j] + g1 * m1[j] + g2 * m2[j] + (j == l ? 1.0f : 0.0f);
    float v = ar[j] * fm;
    srow[j] = v; mx = fmaxf(mx, v);
  }
  red[tid] = mx; __syncthreads();
  for (int s = 128; s > 0; s >>= 1) { if (tid < s) red[tid] = fmaxf(red[tid], red[tid + s]); __syncthreads(); }
  mx = red[0]; __syncthreads();
  float sm = 0;
  for (int j = tid; j < Lv; j += 256) { float e = expf(srow[j] - mx); srow[j] = e; sm += e; }
  red[tid] = sm; __syncthreads();
  for (int s = 128; s > 0; s >>= 1) { if (tid < s) red[tid] += red[tid + s]; __syncthreads(); }
  float inv = 1.0f / red[0];
  __syncthreads();
  _Float16* o16 = (_Float16*)ar;                           // softmax rows sum to 1 -> adjn == adj
  for (int j = tid; j < Lv; j += 256) o16[j] = (_Float16)(srow[j] * inv);
}

// -------- TDM helper: issue tensor_load_to_lds of a 16 x width f16 tile (row stride = strideK) ----
__device__ __forceinline__ void tdm_load_panel(const _Float16* gsrc, unsigned width, unsigned strideK) {
  unsigned long long ga = (unsigned long long)(__UINTPTR_TYPE__)gsrc;
  u32x4 g0;
  g0[0] = 1u;                                           // count=1 (valid user descriptor)
  g0[1] = 0u;                                           // lds_addr = 0 (first shared var)
  g0[2] = (unsigned)(ga & 0xFFFFFFFFu);                 // global_addr[31:0]
  g0[3] = (unsigned)((ga >> 32) & 0x01FFFFFFu) | (2u << 30);  // addr[56:32] | type=2
  i32x8 g1;
  g1[0] = (int)(1u << 16);                              // data_size=1 (2 bytes)
  g1[1] = (int)((width & 0xFFFFu) << 16);               // tensor_dim0[15:0]
  g1[2] = (int)((width >> 16) & 0xFFFFu) | (int)(16u << 16); // td0[31:16] | tensor_dim1=16
  g1[3] = (int)((width & 0xFFFFu) << 16);               // tile_dim0 = width
  g1[4] = 16;                                           // tile_dim1 = 16, tile_dim2 = 0
  g1[5] = (int)strideK;                                 // tensor_dim0_stride[31:0]
  g1[6] = 0;                                            // stride[47:32]=0, dim1_stride lo=0
  g1[7] = 0;
  i32x4 gz4 = {0, 0, 0, 0};
  i32x8 gz8 = {0, 0, 0, 0, 0, 0, 0, 0};
  __builtin_amdgcn_tensor_load_to_lds(g0, g1, gz4, gz4, gz8, 0);  // 6-arg clang-23 form
  __builtin_amdgcn_s_wait_tensorcnt(0);
}

// ---------------- generic f16 GEMM: out = act(A[M,K] @ Wt[N,K]^T + bias) ----------------
// TDM stages the block-shared 16-row A panel into LDS in KCH chunks.
// mode 0: store f16 TRANSPOSED [N,M]; mode 1: gelu + store f16 [M,N]; mode 2: +resid, store f32
__global__ void gemm_kernel(const _Float16* __restrict__ A, const _Float16* __restrict__ WT,
                            const float* __restrict__ bias, int M, int N, int K,
                            int mode, _Float16* __restrict__ out16, float* __restrict__ out32,
                            const float* __restrict__ resid) {
  __shared__ __align__(16) _Float16 As[16 * KCH];        // 16 KB A panel chunk
  int wave = threadIdx.x >> 5;
  int tile = blockIdx.x * 8 + wave;
  int nt = N >> 4;
  int rt = tile / nt, ct = tile % nt;                    // all 8 waves share rt (nt % 8 == 0)
  const _Float16* Ab = A + (size_t)rt * 16 * K;
  const _Float16* Bt = WT + (size_t)(ct * 16) * K;       // B(k,n) = Bt[n*K + k]
  int lane = threadIdx.x & 31;
  v8f acc = {};
  for (int kb = 0; kb < K; kb += KCH) {
    if (wave == 0)
      tdm_load_panel(Ab + kb, KCH, (unsigned)K);         // TDM: global -> LDS, s_wait_tensorcnt
    __syncthreads();
    __builtin_prefetch(Bt + (size_t)(lane & 15) * K + kb + KCH, 0, 1);  // stream next B chunk
    for (int k0 = 0; k0 < KCH; k0 += 32) {
      v16h af = load_a_frag(As + k0, KCH);               // ds reads of staged panel
      v16h bf = load_bT_frag(Bt + kb + k0, K);
      acc = __builtin_amdgcn_wmma_f32_16x16x32_f16(false, af, false, bf, (short)0, acc, false, false);
    }
    __syncthreads();
  }
  int n = lane & 15, hl = lane >> 4;
#pragma unroll
  for (int r = 0; r < 8; ++r) {
    int m = rt * 16 + hl * 8 + r;
    int c = ct * 16 + n;
    float v = acc[r] + bias[c];
    if (mode == 0) {
      out16[(size_t)c * M + m] = (_Float16)v;            // transposed for downstream B use
    } else if (mode == 1) {
      out16[(size_t)m * N + c] = (_Float16)gelu_exact(v);
    } else {
      size_t idx = (size_t)m * N + c;
      out32[idx] = v + resid[idx];
    }
  }
}

// ---------------- aggregation: x1 = x + adj16 @ xp (per head), xp stored transposed ----------------
__global__ void agg_kernel(const float* __restrict__ adj, const _Float16* __restrict__ xpT,
                           const float* __restrict__ x, float* __restrict__ x1) {
  int wave = threadIdx.x >> 5;
  int tile = blockIdx.x * 8 + wave;
  int ct = tile & 3; int t2 = tile >> 2;
  int rt = t2 % LTv; int bh = t2 / LTv;
  int b = bh / Hv, h = bh % Hv;
  const int M2 = Bv * Lv;
  // f16 row overlaid at start of each f32 adj row -> row stride 2*L halves
  const _Float16* Ab = (const _Float16*)(adj + ((size_t)bh * Lv + rt * 16) * Lv);
  const _Float16* Bt = xpT + (size_t)(h * DHv + ct * 16) * M2 + (size_t)b * Lv;  // (k,n)=Bt[n*M2+k]
  int lane = threadIdx.x & 31;
  v8f acc = {};
  for (int k0 = 0; k0 < Lv; k0 += 32) {
    __builtin_prefetch(Bt + (size_t)(lane & 15) * M2 + k0 + 128, 0, 1);
    v16h a  = load_a_frag(Ab + k0, 2 * Lv);
    v16h bf = load_bT_frag(Bt + k0, (size_t)M2);
    acc = __builtin_amdgcn_wmma_f32_16x16x32_f16(false, a, false, bf, (short)0, acc, false, false);
  }
  int n = lane & 15, hl = lane >> 4;
#pragma unroll
  for (int r = 0; r < 8; ++r) {
    int m = rt * 16 + hl * 8 + r;
    int c = h * DHv + ct * 16 + n;
    size_t idx = ((size_t)b * Lv + m) * Dv + c;
    x1[idx] = x[idx] + acc[r];
  }
}

// ---------------- final loss: loss_imp + 0.1*loss_dyn -> d_out[B*L*D] ----------------
__global__ void loss_kernel(const float* __restrict__ sps, const float* __restrict__ ent,
                            float* __restrict__ out) {
  __shared__ float rs[256], rq[256], re[256];
  int tid = threadIdx.x;
  float s = 0, q = 0;
  for (int idx = tid; idx < Lv * 3; idx += 256) {
    float v = 0;
    for (int bh = 0; bh < BHv; ++bh) v += sps[(size_t)bh * Lv * 3 + idx];  // spz[l,j]
    s += v; q += v * v;
  }
  float e = 0;
  for (int i = tid; i < BHv * Lv; i += 256) e += ent[i];
  rs[tid] = s; rq[tid] = q; re[tid] = e; __syncthreads();
  for (int st = 128; st > 0; st >>= 1) {
    if (tid < st) { rs[tid] += rs[tid + st]; rq[tid] += rq[tid + st]; re[tid] += re[tid + st]; }
    __syncthreads();
  }
  if (tid == 0) {
    const float n = (float)(Lv * 3);
    float mean = rs[0] / n;
    float var = (rq[0] - n * mean * mean) / (n - 1.0f);    // ddof=1
    float loss_imp = var / (mean * mean + 1e-10f);
    float loss_dyn = re[0] / (float)(BHv * 3);
    out[(size_t)Bv * Lv * Dv] = loss_imp + 0.1f * loss_dyn;
  }
}

extern "C" void kernel_launch(void* const* d_in, const int* in_sizes, int n_in,
                              void* d_out, int out_size, void* d_ws, size_t ws_size,
                              hipStream_t stream) {
  (void)in_sizes; (void)n_in; (void)out_size; (void)ws_size;
  const float* x     = (const float*)d_in[0];
  const float* masks = (const float*)d_in[1];
  const float* ln1g  = (const float*)d_in[2];
  const float* ln1b  = (const float*)d_in[3];
  const float* wp1   = (const float*)d_in[4];
  const float* bp1   = (const float*)d_in[5];
  const float* wp2   = (const float*)d_in[6];
  const float* bp2   = (const float*)d_in[7];
  const float* wgate = (const float*)d_in[8];
  const float* wgcn  = (const float*)d_in[9];
  const float* bgcn  = (const float*)d_in[10];
  const float* ln2g  = (const float*)d_in[11];
  const float* ln2b  = (const float*)d_in[12];
  const float* wf1   = (const float*)d_in[13];
  const float* bf1   = (const float*)d_in[14];
  const float* wf2   = (const float*)d_in[15];
  const float* bf2   = (const float*)d_in[16];
  float* out = (float*)d_out;

  char* ws = (char*)d_ws;
  size_t off = 0;
  auto take = [&](size_t bytes) -> char* {
    char* p = ws + off;
    off = (off + bytes + 255) & ~(size_t)255;
    return p;
  };
  _Float16* xn16  = (_Float16*)take((size_t)Bv * Lv * Dv * 2);
  _Float16* q16   = (_Float16*)take((size_t)BHv * Lv * DHv * 2);
  _Float16* k16   = (_Float16*)take((size_t)BHv * Lv * DHv * 2);
  float*    adj   = (float*)   take((size_t)BHv * Lv * Lv * 4);    // 127 MB
  float*    gates = (float*)   take((size_t)BHv * Lv * 3 * 4);
  float*    sps   = (float*)   take((size_t)BHv * Lv * 3 * 4);
  float*    ent   = (float*)   take((size_t)BHv * Lv * 4);
  _Float16* xpT   = (_Float16*)take((size_t)Bv * Lv * Dv * 2);     // [D, B*L] transposed
  float*    x1    = (float*)   take((size_t)Bv * Lv * Dv * 4);
  _Float16* xn216 = (_Float16*)take((size_t)Bv * Lv * Dv * 2);
  _Float16* h16   = (_Float16*)take((size_t)Bv * Lv * DFFv * 2);
  _Float16* wp1t  = (_Float16*)take((size_t)DHv * DHv * 2);        // [N,K] transposed weights
  _Float16* wp2t  = (_Float16*)take((size_t)DHv * DHv * 2);
  _Float16* wgcnt = (_Float16*)take((size_t)Dv * Dv * 2);
  _Float16* wf1t  = (_Float16*)take((size_t)Dv * DFFv * 2);
  _Float16* wf2t  = (_Float16*)take((size_t)DFFv * Dv * 2);

  dim3 b256(256);
  cvt_f16_t<<<(DHv * DHv + 255) / 256, b256, 0, stream>>>(wp1, wp1t, DHv, DHv);
  cvt_f16_t<<<(DHv * DHv + 255) / 256, b256, 0, stream>>>(wp2, wp2t, DHv, DHv);
  cvt_f16_t<<<(Dv * Dv + 255) / 256, b256, 0, stream>>>(wgcn, wgcnt, Dv, Dv);
  cvt_f16_t<<<(Dv * DFFv + 255) / 256, b256, 0, stream>>>(wf1, wf1t, Dv, DFFv);
  cvt_f16_t<<<(DFFv * Dv + 255) / 256, b256, 0, stream>>>(wf2, wf2t, DFFv, Dv);

  ln_kernel<<<Bv * Lv, b256, 0, stream>>>(x, ln1g, ln1b, xn16);
  qk_proj_kernel<<<BHv * LTv, b256, 0, stream>>>(xn16, wp1t, bp1, wp2t, bp2, q16, k16);
  adj_gemm_kernel<<<BHv * LTv * (LTv / 8), b256, 0, stream>>>(q16, k16, adj);
  topk_kernel<<<BHv * Lv, b256, 0, stream>>>(adj);
  gate_kernel<<<BHv * Lv, b256, 0, stream>>>(adj, wgate, gates, sps, ent);
  fsm_kernel<<<BHv * Lv, b256, 0, stream>>>(adj, gates, masks);
  gemm_kernel<<<(Bv * Lv / 16) * (Dv / 16) / 8, b256, 0, stream>>>(
      xn16, wgcnt, bgcn, Bv * Lv, Dv, Dv, 0, xpT, nullptr, nullptr);      // -> xp transposed
  agg_kernel<<<BHv * LTv * (DHv / 16) / 8, b256, 0, stream>>>(adj, xpT, x, x1);
  ln_kernel<<<Bv * Lv, b256, 0, stream>>>(x1, ln2g, ln2b, xn216);
  gemm_kernel<<<(Bv * Lv / 16) * (DFFv / 16) / 8, b256, 0, stream>>>(
      xn216, wf1t, bf1, Bv * Lv, DFFv, Dv, 1, h16, nullptr, nullptr);     // gelu FFN1
  gemm_kernel<<<(Bv * Lv / 16) * (Dv / 16) / 8, b256, 0, stream>>>(
      h16, wf2t, bf2, Bv * Lv, Dv, DFFv, 2, nullptr, out, x1);            // FFN2 + residual
  loss_kernel<<<1, b256, 0, stream>>>(sps, ent, out);
}
